// SAICNet_82566451298930
// MI455X (gfx1250) — matrix operand order repaired and checked by reading, not compiled, not run on hardware
//
#include <hip/hip_runtime.h>
#include <hip/hip_bf16.h>

typedef __attribute__((ext_vector_type(16))) _Float16 v16h;
typedef __attribute__((ext_vector_type(8)))  float    v8f;
typedef int v4i_ __attribute__((vector_size(16)));   // matches async-LDS builtin param type

#define N_AG   131072   // total agents
#define N_SC   4096     // scenes
#define T_IN_  20
#define T_OUT_ 30

#if __has_builtin(__builtin_amdgcn_global_load_async_to_lds_b128)
#define HAVE_ASYNC_LDS 1
#endif

// ---- fragment slot helpers (CDNA5 ISA 7.12.2 layouts, 16-bit, wave32) ----
// A-matrix 16x32: lane = m + 16*(k&8?1:0); element e = (k&7) + ((k&16)>>1)
__device__ __forceinline__ int frag_slot(int m, int k) {
    return ((m + ((k & 8) << 1)) << 4) + (k & 7) + ((k & 16) >> 1);
}
// B-matrix 32x16: lane = n + 16*(k&8?1:0); element e = (k&7) + ((k&16)>>1)
__device__ __forceinline__ int b_slot(int n, int k) {
    return ((n + ((k & 8) << 1)) << 4) + (k & 7) + ((k & 16) >> 1);
}

__device__ __forceinline__ v16h load_frag(const _Float16* base, int lane) {
    return *(const v16h*)(base + (lane << 4));
}

__device__ __forceinline__ v8f wmma16(v16h a, v16h b, v8f c) {
    return __builtin_amdgcn_wmma_f32_16x16x32_f16(false, a, false, b, (short)0, c, false, false);
}

// Branch-free activations: native v_tanh_f32 if available, v_exp/v_rcp otherwise.
__device__ __forceinline__ float fast_sigmoid(float x) {
    return __builtin_amdgcn_rcpf(1.0f + __expf(-x));
}
__device__ __forceinline__ float fast_tanh(float x) {
#if __has_builtin(__builtin_amdgcn_tanhf)
    return __builtin_amdgcn_tanhf(x);
#else
    return 1.0f - 2.0f * __builtin_amdgcn_rcpf(1.0f + __expf(2.0f * x));
#endif
}

// =====================================================================
// Kernel 1: input embedding + encoder LSTM (20 steps), 16 agents / tile,
// 16 tiles / block, 4 waves: wave w owns hidden cols [16w,16w+16) x 4 gates.
// B fragments live in VGPRs; hist staged tile-wise via async LDS loads.
// =====================================================================
__global__ __launch_bounds__(128) void encoder_kernel(
    const float* __restrict__ hist, const float* __restrict__ W_ie, const float* __restrict__ b_ie,
    const float* __restrict__ Wih_e, const float* __restrict__ Whh_e,
    const float* __restrict__ bih_e, const float* __restrict__ bhh_e,
    _Float16* __restrict__ hist_enc_h)
{
    __shared__ __align__(32) _Float16 sWih[16 * 512];   // staging: 16 col-tiles, K=32
    __shared__ __align__(32) _Float16 sWhh[32 * 512];   // staging: 16 col-tiles x 2 K-blocks
    __shared__ __align__(32) _Float16 sX[512];          // emb A-fragment (16x32)
    __shared__ __align__(32) _Float16 sH[2 * 512];      // h A-fragments (16x64)
    __shared__ __align__(16) float sHist[T_IN_ * 32];   // staged inputs for current tile
    __shared__ float sWie[96];                          // W_ie rows + b_ie

    const int tid  = threadIdx.x;
    const int wave = tid >> 5;
    const int lane = tid & 31;
    const int coln = lane & 15;
    const int rowb = (lane >> 4) << 3;

    for (int idx = tid; idx < 32 * 256; idx += 128) {
        int k = idx >> 8, col = idx & 255;
        sWih[((col >> 4) << 9) + b_slot(col & 15, k)] = (_Float16)Wih_e[idx];
    }
    for (int idx = tid; idx < 64 * 256; idx += 128) {
        int k = idx >> 8, col = idx & 255;
        sWhh[((((col >> 4) << 1) + (k >> 5)) << 9) + b_slot(col & 15, k & 31)] = (_Float16)Whh_e[idx];
    }
    for (int idx = tid; idx < 96; idx += 128)
        sWie[idx] = (idx < 64) ? W_ie[idx] : b_ie[idx - 64];

    float bias[4];
    #pragma unroll
    for (int g = 0; g < 4; ++g) {
        int j = g * 64 + wave * 16 + coln;
        bias[g] = bih_e[j] + bhh_e[j];
    }
    __syncthreads();

    // hoist all B fragments into VGPRs (invariant over tiles & time)
    v16h bX[4], bH0[4], bH1[4];
    #pragma unroll
    for (int g = 0; g < 4; ++g) {
        const int ct = g * 4 + wave;
        bX[g]  = load_frag(sWih + (ct << 9), lane);
        bH0[g] = load_frag(sWhh + ((ct << 1) << 9), lane);
        bH1[g] = load_frag(sWhh + (((ct << 1) + 1) << 9), lane);
    }

    for (int tile = 0; tile < 16; ++tile) {
        const int agent0 = (blockIdx.x * 16 + tile) * 16;
        // ---- stage hist[0..19][agent0..agent0+15][2] into LDS ----
#ifdef HAVE_ASYNC_LDS
        for (int i = tid; i < 160; i += 128) {       // 160 x b128 = 640 floats
            int t = i >> 3, o = i & 7;
            const float* gp = hist + (size_t)t * (N_AG * 2) + (size_t)agent0 * 2 + o * 4;
            __builtin_amdgcn_global_load_async_to_lds_b128(
                (v4i_*)gp, (v4i_*)&sHist[t * 32 + o * 4], 0, 0);
        }
#if __has_builtin(__builtin_amdgcn_s_wait_asynccnt)
        __builtin_amdgcn_s_wait_asynccnt(0);
#else
        asm volatile("s_wait_asynccnt 0x0" ::: "memory");
#endif
#else
        for (int i = tid; i < T_IN_ * 32; i += 128) {
            int t = i >> 5, o = i & 31;
            sHist[i] = hist[(size_t)t * (N_AG * 2) + (size_t)agent0 * 2 + o];
        }
#endif
        for (int i = tid; i < 1024; i += 128) sH[i] = (_Float16)0.f;
        v8f c = {};
        float hk[8];
        #pragma unroll
        for (int r = 0; r < 8; ++r) hk[r] = 0.f;
        __syncthreads();

        for (int t = 0; t < T_IN_; ++t) {
            // emb = relu(x @ W_ie + b_ie), written as A-fragment
            for (int idx = tid; idx < 512; idx += 128) {
                int m = idx & 15, k = idx >> 4;
                float x0 = sHist[t * 32 + m * 2], x1 = sHist[t * 32 + m * 2 + 1];
                float v = x0 * sWie[k] + x1 * sWie[32 + k] + sWie[64 + k];
                sX[frag_slot(m, k)] = (_Float16)fmaxf(v, 0.f);
            }
            __syncthreads();

            v16h xa  = load_frag(sX, lane);
            v16h ha0 = load_frag(sH, lane);
            v16h ha1 = load_frag(sH + 512, lane);
            v8f acc[4];
            #pragma unroll
            for (int g = 0; g < 4; ++g) {
                v8f a;
                #pragma unroll
                for (int r = 0; r < 8; ++r) a[r] = bias[g];
                a = wmma16(xa,  bX[g],  a);
                a = wmma16(ha0, bH0[g], a);
                a = wmma16(ha1, bH1[g], a);
                acc[g] = a;
            }
            float hn[8];
            #pragma unroll
            for (int r = 0; r < 8; ++r) {
                float ig = fast_sigmoid(acc[0][r]);
                float fg = fast_sigmoid(acc[1][r]);
                float gg = fast_tanh(acc[2][r]);
                float og = fast_sigmoid(acc[3][r]);
                float cc = fg * c[r] + ig * gg;
                c[r] = cc;
                hn[r] = og * fast_tanh(cc);
                hk[r] = hn[r];
            }
            __syncthreads();   // all fragment reads of sH done
            #pragma unroll
            for (int r = 0; r < 8; ++r) {
                int m = r + rowb;
                int j = wave * 16 + coln;
                sH[((j >> 5) << 9) + frag_slot(m, j & 31)] = (_Float16)hn[r];
            }
            __syncthreads();
        }
        #pragma unroll
        for (int r = 0; r < 8; ++r) {
            int m = r + rowb;
            hist_enc_h[(size_t)(agent0 + m) * 64 + wave * 16 + coln] = (_Float16)hk[r];
        }
    }
}

// =====================================================================
// Kernel 2a: social pooling per scene -> enc[S,128] (f16 row-major)
// =====================================================================
__global__ __launch_bounds__(128) void pool_kernel(
    const _Float16* __restrict__ hist_enc_h, const float* __restrict__ hist_pos,
    const float* __restrict__ Wse, const float* __restrict__ bse,
    const float* __restrict__ Wmp, const float* __restrict__ bmp,
    _Float16* __restrict__ enc_h)
{
    __shared__ __align__(32) _Float16 sWmp[16 * 512];   // 4 ct x 4 kb (staging)
    __shared__ __align__(32) _Float16 sA[8 * 512];      // 2 row-tiles x 4 kb
    __shared__ float sPos[64];
    __shared__ float sWse[192];                         // Wse rows + bse

    const int tid = threadIdx.x, wave = tid >> 5, lane = tid & 31, coln = lane & 15;

    for (int idx = tid; idx < 128 * 64; idx += 128) {
        int k = idx >> 6, col = idx & 63;
        sWmp[(((col >> 4) << 2) + (k >> 5)) * 512 + b_slot(col & 15, k & 31)] = (_Float16)Wmp[idx];
    }
    for (int idx = tid; idx < 192; idx += 128)
        sWse[idx] = (idx < 128) ? Wse[idx] : bse[idx - 128];
    const float bmpv = bmp[wave * 16 + coln];
    __syncthreads();

    v16h bW[4];
    #pragma unroll
    for (int kb = 0; kb < 4; ++kb) bW[kb] = load_frag(sWmp + ((wave * 4 + kb) << 9), lane);

    for (int si = 0; si < 16; ++si) {
        const int s = blockIdx.x * 16 + si;
        if (tid < 64) sPos[tid] = hist_pos[(size_t)s * 64 + tid];
        __syncthreads();
        // A = [H | rel_emb]  (32 x 128)
        for (int idx = tid; idx < 2048; idx += 128) {
            int p = idx >> 6, j = idx & 63;
            sA[(((p >> 4) << 2) + (j >> 5)) * 512 + frag_slot(p & 15, j & 31)] =
                hist_enc_h[((size_t)s * 32 + p) * 64 + j];
        }
        for (int idx = tid; idx < 2048; idx += 128) {
            int p = idx >> 6, q = idx & 63;
            float r0 = sPos[p * 2]     - sPos[0];
            float r1 = sPos[p * 2 + 1] - sPos[1];
            float v = fmaxf(r0 * sWse[q] + r1 * sWse[64 + q] + sWse[128 + q], 0.f);
            sA[(((p >> 4) << 2) + 2 + (q >> 5)) * 512 + frag_slot(p & 15, q & 31)] = (_Float16)v;
        }
        __syncthreads();

        float colmax = 0.f;   // relu then max == max with 0 floor
        #pragma unroll
        for (int rt = 0; rt < 2; ++rt) {
            v8f acc;
            #pragma unroll
            for (int r = 0; r < 8; ++r) acc[r] = bmpv;
            #pragma unroll
            for (int kb = 0; kb < 4; ++kb)
                acc = wmma16(load_frag(sA + ((rt * 4 + kb) << 9), lane), bW[kb], acc);
            #pragma unroll
            for (int r = 0; r < 8; ++r) colmax = fmaxf(colmax, acc[r]);
        }
        colmax = fmaxf(colmax, __shfl_xor(colmax, 16));
        if (lane < 16) enc_h[(size_t)s * 128 + wave * 16 + coln] = (_Float16)colmax;
        if (tid < 64)  enc_h[(size_t)s * 128 + 64 + tid] = hist_enc_h[(size_t)s * 32 * 64 + tid];
        __syncthreads();
    }
}

// =====================================================================
// Kernel 2b: gx = enc @ Wih_d + bih_d + bhh_d   (M=4096, K=128, N=256)
// =====================================================================
__global__ __launch_bounds__(128) void gx_kernel(
    const _Float16* __restrict__ enc_h, const float* __restrict__ Wih_d,
    const float* __restrict__ bih_d, const float* __restrict__ bhh_d,
    float* __restrict__ gx)
{
    __shared__ __align__(32) _Float16 sA[4 * 512];
    __shared__ __align__(32) _Float16 sB[16 * 512];   // 4 ct (one per wave) x 4 kb
    const int tid = threadIdx.x, wave = tid >> 5, lane = tid & 31, coln = lane & 15;
    const int s0 = blockIdx.x * 16;
    const int rowb = (lane >> 4) << 3;

    for (int idx = tid; idx < 16 * 128; idx += 128) {
        int m = idx & 15, k = idx >> 4;
        sA[((k >> 5) << 9) + frag_slot(m, k & 31)] = enc_h[(size_t)(s0 + m) * 128 + k];
    }
    for (int q = 0; q < 4; ++q) {
        __syncthreads();
        for (int idx = tid; idx < 128 * 64; idx += 128) {
            int k = idx >> 6, c = idx & 63;
            sB[(((c >> 4) << 2) + (k >> 5)) * 512 + b_slot(c & 15, k & 31)] =
                (_Float16)Wih_d[(size_t)k * 256 + q * 64 + c];
        }
        __syncthreads();
        const int col = (q * 4 + wave) * 16 + coln;
        const float bias = bih_d[col] + bhh_d[col];
        v8f acc;
        #pragma unroll
        for (int r = 0; r < 8; ++r) acc[r] = bias;
        #pragma unroll
        for (int kb = 0; kb < 4; ++kb)
            acc = wmma16(load_frag(sA + (kb << 9), lane),
                         load_frag(sB + ((wave * 4 + kb) << 9), lane), acc);
        #pragma unroll
        for (int r = 0; r < 8; ++r)
            gx[(size_t)(s0 + r + rowb) * 256 + col] = acc[r];
    }
}

// =====================================================================
// Kernel 3: decoder LSTM (30 steps) + readout, 16 scenes / block
// =====================================================================
__global__ __launch_bounds__(128) void dec_kernel(
    const float* __restrict__ gx, const float* __restrict__ Whh_d,
    const float* __restrict__ Wop, const float* __restrict__ bop,
    float* __restrict__ out)
{
    __shared__ __align__(32) _Float16 sWhh[32 * 512];
    __shared__ __align__(32) _Float16 sH[2 * 512];
    __shared__ __align__(32) _Float16 sHrow[16 * 64];
    __shared__ float sWop[130];

    const int tid = threadIdx.x, wave = tid >> 5, lane = tid & 31, coln = lane & 15;
    const int s0 = blockIdx.x * 16;
    const int rowb = (lane >> 4) << 3;

    for (int idx = tid; idx < 64 * 256; idx += 128) {
        int k = idx >> 8, col = idx & 255;
        sWhh[((((col >> 4) << 1) + (k >> 5)) << 9) + b_slot(col & 15, k & 31)] = (_Float16)Whh_d[idx];
    }
    for (int idx = tid; idx < 130; idx += 128) sWop[idx] = (idx < 128) ? Wop[idx] : bop[idx - 128];
    for (int i = tid; i < 1024; i += 128) sH[i] = (_Float16)0.f;

    float gxr[4][8];
    #pragma unroll
    for (int g = 0; g < 4; ++g)
        #pragma unroll
        for (int r = 0; r < 8; ++r)
            gxr[g][r] = gx[(size_t)(s0 + r + rowb) * 256 + g * 64 + wave * 16 + coln];
    v8f c = {};
    __syncthreads();

    v16h bH0[4], bH1[4];
    #pragma unroll
    for (int g = 0; g < 4; ++g) {
        const int ct = g * 4 + wave;
        bH0[g] = load_frag(sWhh + ((ct << 1) << 9), lane);
        bH1[g] = load_frag(sWhh + (((ct << 1) + 1) << 9), lane);
    }

    for (int t = 0; t < T_OUT_; ++t) {
        v16h h0 = load_frag(sH, lane), h1 = load_frag(sH + 512, lane);
        v8f acc[4];
        #pragma unroll
        for (int g = 0; g < 4; ++g) {
            v8f a;
            #pragma unroll
            for (int r = 0; r < 8; ++r) a[r] = gxr[g][r];
            a = wmma16(h0, bH0[g], a);
            a = wmma16(h1, bH1[g], a);
            acc[g] = a;
        }
        float hn[8];
        #pragma unroll
        for (int r = 0; r < 8; ++r) {
            float ig = fast_sigmoid(acc[0][r]);
            float fg = fast_sigmoid(acc[1][r]);
            float gg = fast_tanh(acc[2][r]);
            float og = fast_sigmoid(acc[3][r]);
            float cc = fg * c[r] + ig * gg;
            c[r] = cc;
            hn[r] = og * fast_tanh(cc);
        }
        __syncthreads();
        #pragma unroll
        for (int r = 0; r < 8; ++r) {
            int m = r + rowb;
            int j = wave * 16 + coln;
            _Float16 hv = (_Float16)hn[r];
            sH[((j >> 5) << 9) + frag_slot(m, j & 31)] = hv;
            sHrow[m * 64 + j] = hv;
        }
        __syncthreads();
        if (tid < 32) {   // whole wave 0 -> no intra-wave divergence
            int m = tid >> 1, d = tid & 1;
            float sum = sWop[128 + d];
            #pragma unroll
            for (int j = 0; j < 64; ++j) sum += (float)sHrow[m * 64 + j] * sWop[j * 2 + d];
            out[(size_t)t * (N_SC * 2) + (size_t)(s0 + m) * 2 + d] = sum;
        }
    }
}

extern "C" void kernel_launch(void* const* d_in, const int* in_sizes, int n_in,
                              void* d_out, int out_size, void* d_ws, size_t ws_size,
                              hipStream_t stream) {
    (void)in_sizes; (void)n_in; (void)out_size; (void)ws_size;
    const float* hist     = (const float*)d_in[0];
    const float* hist_pos = (const float*)d_in[1];
    const float* W_ie  = (const float*)d_in[2];
    const float* b_ie  = (const float*)d_in[3];
    const float* Wih_e = (const float*)d_in[4];
    const float* Whh_e = (const float*)d_in[5];
    const float* bih_e = (const float*)d_in[6];
    const float* bhh_e = (const float*)d_in[7];
    const float* Wse   = (const float*)d_in[8];
    const float* bse   = (const float*)d_in[9];
    const float* Wmp   = (const float*)d_in[10];
    const float* bmp   = (const float*)d_in[11];
    const float* Wih_d = (const float*)d_in[12];
    const float* Whh_d = (const float*)d_in[13];
    const float* bih_d = (const float*)d_in[14];
    const float* bhh_d = (const float*)d_in[15];
    const float* Wop   = (const float*)d_in[16];
    const float* bop   = (const float*)d_in[17];

    char* ws = (char*)d_ws;
    _Float16* hist_enc_h = (_Float16*)ws;                           // 16 MB
    _Float16* enc_h      = (_Float16*)(ws + 16ull * 1024 * 1024);   // 1 MB
    float*    gx         = (float*)   (ws + 17ull * 1024 * 1024);   // 4 MB
    float*    out        = (float*)d_out;

    encoder_kernel<<<512, 128, 0, stream>>>(hist, W_ie, b_ie, Wih_e, Whh_e, bih_e, bhh_e, hist_enc_h);
    pool_kernel  <<<256, 128, 0, stream>>>(hist_enc_h, hist_pos, Wse, bse, Wmp, bmp, enc_h);
    gx_kernel    <<<256, 128, 0, stream>>>(enc_h, Wih_d, bih_d, bhh_d, gx);
    dec_kernel   <<<256, 128, 0, stream>>>(gx, Whh_d, Wop, bop, out);
}
